// LNO_Jamba_Layer_1_75505525064750
// MI455X (gfx1250) — matrix operand order repaired and checked by reading, not compiled, not run on hardware
//
#include <hip/hip_runtime.h>
#include <hip/hip_bf16.h>
#include <math.h>

// Problem constants: B=4, NX=512, NT=128, W=64, M=16
#define BDIM 4
#define NXD  512
#define NTD  128
#define WD   64
#define MD   16
#define BT   (BDIM*NTD)      // 512
#define ROWS (BT*WD)         // 32768
#define PI_F 3.14159265358979f

typedef __attribute__((ext_vector_type(16))) _Float16 v16h;
typedef __attribute__((ext_vector_type(8)))  float    v8f;
typedef __attribute__((ext_vector_type(4)))  unsigned int u32x4;
typedef __attribute__((ext_vector_type(8)))  int i32x8;
typedef __attribute__((ext_vector_type(4)))  int i32x4;

#if defined(__has_builtin)
#if __has_builtin(__builtin_amdgcn_tensor_load_to_lds) && __has_builtin(__builtin_amdgcn_s_wait_tensorcnt)
#define HAVE_TDM 1
#endif
#endif
#if __has_include(<hip/amd_detail/amd_gfx1250_TDM.h>)
#define TDM_6ARG 1
#endif

__device__ __forceinline__ v8f wmma_f16(v16h a, v16h b, v8f c) {
  return __builtin_amdgcn_wmma_f32_16x16x32_f16(false, a, false, b, (short)0, c, false, false);
}

// ---- swizzle index helpers (WMMA 16x16x32 f16 operands, contiguous k = 16*half + j order used here)
// A tile 16(M)x32(K): lane = half*16+row holds j=0..15 with K = half*16+j
__device__ __forceinline__ int a_swz(int row, int kin) {   // -> element index in [0,512)
  int h = kin >> 4;
  int j = kin & 15;
  return (h * 16 + row) * 16 + j;
}
// B tile 32(K)x16(N): lane = half*16+col holds j=0..15 with K = half*16+j
__device__ __forceinline__ int b_swz(int kin, int n) {
  int h2 = kin >> 4, j = kin & 15;
  return (h2 * 16 + n) * 16 + j;
}

// ---- K1: transpose x[B,NX,NT,W] -> xs[BT,W,NX] + instance-norm copy ------
__global__ void k_pack_norm(const float* __restrict__ x, float* __restrict__ xs,
                            float* __restrict__ xsn) {
  __shared__ float red[256];
  int row = blockIdx.x;            // bt*W + i
  int bt = row >> 6, i = row & 63;
  int b = bt / NTD, t = bt % NTD;
  int n0 = threadIdx.x, n1 = threadIdx.x + 256;
  float v0 = x[(((size_t)b * NXD + n0) * NTD + t) * WD + i];
  float v1 = x[(((size_t)b * NXD + n1) * NTD + t) * WD + i];
  red[threadIdx.x] = v0 + v1; __syncthreads();
  for (int s = 128; s > 0; s >>= 1) { if (threadIdx.x < s) red[threadIdx.x] += red[threadIdx.x + s]; __syncthreads(); }
  float mu = red[0] * (1.f / 512.f); __syncthreads();
  float d0 = v0 - mu, d1 = v1 - mu;
  red[threadIdx.x] = d0 * d0 + d1 * d1; __syncthreads();
  for (int s = 128; s > 0; s >>= 1) { if (threadIdx.x < s) red[threadIdx.x] += red[threadIdx.x + s]; __syncthreads(); }
  float rs = rsqrtf(red[0] * (1.f / 512.f) + 1e-5f);
  size_t base = (size_t)row * NXD;
  xs[base + n0] = v0;        xs[base + n1] = v1;
  xsn[base + n0] = d0 * rs;  xsn[base + n1] = d1 * rs;
}

// ---- K2: DFT basis in swizzled B-tile layout: [nt=32][kt=16][512] --------
__global__ void k_gen_dft(_Float16* __restrict__ Cm, _Float16* __restrict__ Sm) {
  int idx = blockIdx.x * blockDim.x + threadIdx.x;
  if (idx >= NXD * NXD) return;
  int nt = idx >> 13, rem = idx & 8191;
  int kt = rem >> 9, le = rem & 511;
  int lane = le >> 4, j = le & 15;
  int n = lane & 15, h2 = lane >> 4;
  int k = kt * 32 + h2 * 16 + j;        // DFT input index
  int col = nt * 16 + n;                // frequency
  float ang = 2.f * PI_F * (float)((k * col) & 511) * (1.f / 512.f);
  Cm[idx] = (_Float16)cosf(ang);
  Sm[idx] = (_Float16)sinf(ang);
}

// ---- K3: WMMA GEMM (M=32768, K=512, N=512), TDM-staged A panel -----------
// MODE 0: D f32 row-major, D = alpha*A@B + beta*D
// MODE 1: f16 output scattered into frequency-major swizzled alpha tiles
template <int MODE>
__global__ void k_gemm(const float* __restrict__ A, const _Float16* __restrict__ Bsw,
                       float* __restrict__ D, _Float16* __restrict__ Dh,
                       float alphaS, float betaS) {
  __shared__ float As[16 * 512];                       // raw A panel (LDS offset 0)
  __shared__ __align__(32) _Float16 As2[16 * 512];     // swizzled f16 A tiles
  int mt = blockIdx.x;
  const float* Arow = A + (size_t)mt * 16 * 512;
#if defined(HAVE_TDM)
  if (threadIdx.x < 32) {                              // wave 0 drives the TDM
    unsigned long long ga = (unsigned long long)(uintptr_t)Arow & 0x1FFFFFFFFFFFFFFull;
    u32x4 g0;
    g0[0] = 1u;                                        // count=1 (valid descriptor)
    g0[1] = 0u;                                        // lds_addr = offset of As
    g0[2] = (unsigned)ga;                              // global_addr[31:0]
    g0[3] = (unsigned)((ga >> 32) & 0x1FFFFFFu) | (2u << 30); // addr[56:32] | type=2
    unsigned long long dim0 = 16ull * 512ull;          // 1-D tile, 4B elements
    unsigned long long q0 = (2ull << 16) | ((dim0 & 0xFFFFull) << 48);
    unsigned long long q1 = (dim0 >> 16) | (1ull << 16) | ((dim0 & 0xFFFFull) << 48);
    unsigned long long q2 = (dim0 & 0xFFFFFFFFull) << 32;
    unsigned long long q3 = 0ull;
    i32x8 g1;
    g1[0] = (int)q0; g1[1] = (int)(q0 >> 32);
    g1[2] = (int)q1; g1[3] = (int)(q1 >> 32);
    g1[4] = (int)q2; g1[5] = (int)(q2 >> 32);
    g1[6] = (int)q3; g1[7] = (int)(q3 >> 32);
    i32x4 z4 = {0, 0, 0, 0};
#if defined(TDM_6ARG)
    i32x8 z8 = {0, 0, 0, 0, 0, 0, 0, 0};
    __builtin_amdgcn_tensor_load_to_lds(g0, g1, z4, z4, z8, 0);
#else
    __builtin_amdgcn_tensor_load_to_lds(g0, g1, z4, z4, 0);
#endif
    __builtin_amdgcn_s_wait_tensorcnt(0);
  }
  __syncthreads();
#else
  for (int e = threadIdx.x; e < 16 * 512; e += blockDim.x) As[e] = Arow[e];
  __syncthreads();
#endif
  // swizzle raw f32 panel -> f16 WMMA-A tiles
  for (int e = threadIdx.x; e < 16 * 512; e += blockDim.x) {
    int row = e >> 9, k = e & 511;
    int kt = k >> 5, kin = k & 31;
    As2[kt * 512 + a_swz(row, kin)] = (_Float16)As[row * 512 + k];
  }
  __syncthreads();
  int wave = threadIdx.x >> 5, lane = threadIdx.x & 31;
  int nt = blockIdx.y * 8 + wave;
  const _Float16* Btiles = Bsw + (size_t)nt * 16 * 512;
  v8f acc = {};
  for (int kt = 0; kt < 16; ++kt) {
    v16h a = *(const v16h*)(As2 + (size_t)kt * 512 + lane * 16);
    v16h b = *(const v16h*)(Btiles + (size_t)kt * 512 + lane * 16);
    if (kt < 15) __builtin_prefetch(Btiles + (size_t)(kt + 1) * 512, 0, 1);
    acc = wmma_f16(a, b, acc);
  }
  int n = lane & 15, half = lane >> 4;
  if (MODE == 0) {
    float* drow = D + (size_t)mt * 16 * 512 + nt * 16;
#pragma unroll
    for (int r = 0; r < 8; ++r) {
      int mm = half * 8 + r;
      float v = alphaS * acc[r] + betaS * drow[(size_t)mm * 512 + n];
      drow[(size_t)mm * 512 + n] = v;
    }
  } else {
    // scatter into alpha tiles: [x][bmt=32][kt=2][512] f16
    int x = nt * 16 + n;
#pragma unroll
    for (int r = 0; r < 8; ++r) {
      int rg = mt * 16 + half * 8 + r;   // = b*64 + i
      int bq = rg >> 6, iq = rg & 63;
      size_t dest = (((size_t)x * 32 + (bq >> 4)) * 2 + (iq >> 5)) * 512 + a_swz(bq & 15, iq & 31);
      Dh[dest] = (_Float16)(alphaS * acc[r]);
    }
  }
}

// ---- K4: or1[b,o,x], per-frequency complex GEMM with LDS B tiles ---------
__global__ void k_or1(const _Float16* __restrict__ afr, const _Float16* __restrict__ afi,
                      const float* __restrict__ pre, const float* __restrict__ pim,
                      const float* __restrict__ rre, const float* __restrict__ rim,
                      const float* __restrict__ gridp,
                      float* __restrict__ o1re, float* __restrict__ o1im) {
  __shared__ __align__(32) _Float16 Hr_sw[8 * 512], Hi_sw[8 * 512], Hin_sw[8 * 512];
  int x = blockIdx.x;
  float dt = gridp[1] - gridp[0];
  int kkf = (x < 256) ? x : x - 512;
  float L = 2.f * PI_F * (float)kkf / (512.f * dt);
  for (int e = threadIdx.x; e < 4096; e += 256) {      // Hsum[i][o] = sum_m Hw
    int i = e >> 6, o = e & 63;
    float hr = 0.f, hi = 0.f;
    int base = e * MD;
#pragma unroll
    for (int m = 0; m < MD; ++m) {
      int idx = base + m;
      float dr = -pre[idx], di = L - pim[idx];
      float inv = 1.f / (dr * dr + di * di);
      float nr = rre[idx], ni = rim[idx];
      hr += (nr * dr + ni * di) * inv;
      hi += (ni * dr - nr * di) * inv;
    }
    int dest = ((o >> 4) * 2 + (i >> 5)) * 512 + b_swz(i & 31, o & 15);
    Hr_sw[dest] = (_Float16)hr;
    Hi_sw[dest] = (_Float16)hi;
    Hin_sw[dest] = (_Float16)(-hi);
  }
  __syncthreads();
  int wave = threadIdx.x >> 5, lane = threadIdx.x & 31;
  for (int job = wave; job < 128; job += 8) {          // 32 b-tiles x 4 o-tiles
    int mt = job >> 2, nt = job & 3;
    const _Float16* abase_r = afr + (((size_t)x * 32 + mt) * 2) * 512;
    const _Float16* abase_i = afi + (((size_t)x * 32 + mt) * 2) * 512;
    v8f cre = {}, cim = {};
#pragma unroll
    for (int kt = 0; kt < 2; ++kt) {
      v16h ar = *(const v16h*)(abase_r + (size_t)kt * 512 + lane * 16);
      v16h ai = *(const v16h*)(abase_i + (size_t)kt * 512 + lane * 16);
      v16h br = *(const v16h*)(Hr_sw + (nt * 2 + kt) * 512 + lane * 16);
      v16h bi = *(const v16h*)(Hi_sw + (nt * 2 + kt) * 512 + lane * 16);
      v16h bin = *(const v16h*)(Hin_sw + (nt * 2 + kt) * 512 + lane * 16);
      cre = wmma_f16(ar, br, cre);  cre = wmma_f16(ai, bin, cre);
      cim = wmma_f16(ar, bi, cim);  cim = wmma_f16(ai, br, cim);
    }
    int n = lane & 15, half = lane >> 4;
#pragma unroll
    for (int r = 0; r < 8; ++r) {
      int mm = half * 8 + r;
      size_t off = ((size_t)(mt * 16 + mm) * 64 + nt * 16 + n) * 512 + x;
      o1re[off] = cre[r]; o1im[off] = cim[r];
    }
  }
}

// ---- K5: or2[b,(o,m)] = -sum_{i,x} alpha*Hw, B generated on the fly ------
__global__ void k_or2(const _Float16* __restrict__ afr, const _Float16* __restrict__ afi,
                      const float* __restrict__ pre, const float* __restrict__ pim,
                      const float* __restrict__ rre, const float* __restrict__ rim,
                      const float* __restrict__ gridp,
                      float* __restrict__ o2re, float* __restrict__ o2im) {
  int mt = blockIdx.x;
  int wave = threadIdx.x >> 5, lane = threadIdx.x & 31;
  int nt = blockIdx.y * 8 + wave;
  int n = lane & 15, half = lane >> 4;
  int om = nt * 16 + n, o = om >> 4, m = om & 15;
  float dt = gridp[1] - gridp[0];
  v8f cre = {}, cim = {};
  for (int x = 0; x < 512; ++x) {
    int kkf = (x < 256) ? x : x - 512;
    float L = 2.f * PI_F * (float)kkf / (512.f * dt);
    const _Float16* abase_r = afr + (((size_t)x * 32 + mt) * 2) * 512;
    const _Float16* abase_i = afi + (((size_t)x * 32 + mt) * 2) * 512;
#pragma unroll
    for (int kt = 0; kt < 2; ++kt) {
      v16h ar = *(const v16h*)(abase_r + (size_t)kt * 512 + lane * 16);
      v16h ai = *(const v16h*)(abase_i + (size_t)kt * 512 + lane * 16);
      v16h br, bi, bin;                                // Hw column for this lane
#pragma unroll
      for (int j = 0; j < 16; ++j) {
        int i = kt * 32 + half * 16 + j;
        int idx = (i * 64 + o) * 16 + m;
        float dr = -pre[idx], di = L - pim[idx];
        float inv = 1.f / (dr * dr + di * di);
        float nr = rre[idx], ni = rim[idx];
        float hr = (nr * dr + ni * di) * inv;
        float hi = (ni * dr - nr * di) * inv;
        br[j] = (_Float16)hr; bi[j] = (_Float16)hi; bin[j] = (_Float16)(-hi);
      }
      cre = wmma_f16(ar, br, cre);  cre = wmma_f16(ai, bin, cre);
      cim = wmma_f16(ar, bi, cim);  cim = wmma_f16(ai, br, cim);
    }
  }
#pragma unroll
  for (int r = 0; r < 8; ++r) {                        // negate on store
    int mm = half * 8 + r;
    size_t off = (size_t)(mt * 16 + mm) * 1024 + nt * 16 + n;
    o2re[off] = -cre[r]; o2im[off] = -cim[r];
  }
}

// ---- K5b: re-emit or2 as swizzled f16 A tiles (im pre-negated) -----------
__global__ void k_prep_or2(const float* __restrict__ o2re, const float* __restrict__ o2im,
                           _Float16* __restrict__ swr, _Float16* __restrict__ swi) {
  int idx = blockIdx.x * blockDim.x + threadIdx.x;     // 512*1024
  int b = idx >> 10, c = idx & 1023;
  size_t dest = ((size_t)(b >> 4) * 32 + (c >> 5)) * 512 + a_swz(b & 15, c & 31);
  swr[dest] = (_Float16)o2re[idx];
  swi[dest] = (_Float16)(-o2im[idx]);
}

// ---- K6: x12[b,o,z] += Re(or2 @ exp(pole*grid)) / NX ---------------------
__global__ void k_x2(const _Float16* __restrict__ swr, const _Float16* __restrict__ swi,
                     const float* __restrict__ pre, const float* __restrict__ pim,
                     const float* __restrict__ gridp, float* __restrict__ x12) {
  __shared__ __align__(32) _Float16 Bre_s[32 * 512], Bim_s[32 * 512];   // 64KB
  int o = blockIdx.x, zt = blockIdx.y;
  for (int e = threadIdx.x; e < 16384; e += 256) {     // build term2 tiles once
    int kt = e >> 9, le = e & 511;
    int lpos = le >> 4, j = le & 15;
    int n = lpos & 15, h2 = lpos >> 4;
    int kk = kt * 32 + h2 * 16 + j;
    int i = kk >> 4, m = kk & 15;
    int idx = (i * 64 + o) * 16 + m;
    float g = gridp[zt * 16 + n];
    float er = expf(pre[idx] * g);
    float ph = pim[idx] * g;
    Bre_s[e] = (_Float16)(er * cosf(ph));
    Bim_s[e] = (_Float16)(er * sinf(ph));
  }
  __syncthreads();
  int wave = threadIdx.x >> 5, lane = threadIdx.x & 31;
  int n = lane & 15, half = lane >> 4;
  for (int mt = wave; mt < 32; mt += 8) {
    v8f cre = {};
    for (int kt = 0; kt < 32; ++kt) {
      v16h ar = *(const v16h*)(swr + ((size_t)mt * 32 + kt) * 512 + lane * 16);
      v16h ai = *(const v16h*)(swi + ((size_t)mt * 32 + kt) * 512 + lane * 16);
      v16h br = *(const v16h*)(Bre_s + (size_t)kt * 512 + lane * 16);
      v16h bi = *(const v16h*)(Bim_s + (size_t)kt * 512 + lane * 16);
      cre = wmma_f16(ar, br, cre);                     // Re = Ar@Br + (-Ai)@Bi
      cre = wmma_f16(ai, bi, cre);
    }
#pragma unroll
    for (int r = 0; r < 8; ++r) {
      int mm = half * 8 + r;
      size_t off = ((size_t)(mt * 16 + mm) * 64 + o) * 512 + zt * 16 + n;
      x12[off] += cre[r] * (1.f / 512.f);
    }
  }
}

// ---- K7: instance-norm(x12) + 1x1 conv(xs) -> sin -> GRU layout ----------
__global__ void k_ncs(const float* __restrict__ x12, const float* __restrict__ xs,
                      const float* __restrict__ convw, const float* __restrict__ convb,
                      float* __restrict__ hgru) {
  __shared__ float red[256];
  int row = blockIdx.x; int b = row >> 6, o = row & 63;
  const float* xr = x12 + (size_t)row * 512;
  int z0 = threadIdx.x, z1 = threadIdx.x + 256;
  float v0 = xr[z0], v1 = xr[z1];
  red[threadIdx.x] = v0 + v1; __syncthreads();
  for (int s = 128; s > 0; s >>= 1) { if (threadIdx.x < s) red[threadIdx.x] += red[threadIdx.x + s]; __syncthreads(); }
  float mu = red[0] * (1.f / 512.f); __syncthreads();
  float d0 = v0 - mu, d1 = v1 - mu;
  red[threadIdx.x] = d0 * d0 + d1 * d1; __syncthreads();
  for (int s = 128; s > 0; s >>= 1) { if (threadIdx.x < s) red[threadIdx.x] += red[threadIdx.x + s]; __syncthreads(); }
  float rs = rsqrtf(red[0] * (1.f / 512.f) + 1e-5f);
  float acc0 = convb[o], acc1 = acc0;
  const float* wrow = convw + o * 64;
  const float* xsb = xs + (size_t)b * 64 * 512;
  for (int i = 0; i < 64; ++i) {
    float w = wrow[i];
    acc0 += w * xsb[(size_t)i * 512 + z0];
    acc1 += w * xsb[(size_t)i * 512 + z1];
  }
  float h0 = sinf(d0 * rs + acc0), h1 = sinf(d1 * rs + acc1);
  int borig = b >> 7, t = b & 127;
  hgru[(((size_t)borig * 512 + z0) * 128 + t) * 64 + o] = h0;
  hgru[(((size_t)borig * 512 + z1) * 128 + t) * 64 + o] = h1;
}

// ---- K8: GRU scan, one wave / 16 sequences, swizzled LDS operands --------
__global__ void k_gru(const float* __restrict__ hin, const float* __restrict__ Wx,
                      const float* __restrict__ Wh, const float* __restrict__ bx,
                      const float* __restrict__ bh, float* __restrict__ hs) {
  __shared__ __align__(32) _Float16 Bx_sw[12 * 2 * 512], Bh_sw[12 * 2 * 512];
  __shared__ float gx_s[16 * 192], gh_s[16 * 192], hf[16 * 64];
  __shared__ __align__(32) _Float16 x16[2 * 512], h16[2 * 512];
  int rb = blockIdx.x * 16;
  int lane = threadIdx.x;
  for (int e = lane; e < 12288; e += 32) {             // swizzle weights once
    int nt = e >> 10, rem = e & 1023, kt = rem >> 9, le = rem & 511;
    int l2 = le >> 4, j = le & 15;
    int n = l2 & 15, h2 = l2 >> 4;
    int k = kt * 32 + h2 * 16 + j;
    int g = nt * 16 + n;
    Bx_sw[e] = (_Float16)Wx[g * 64 + k];
    Bh_sw[e] = (_Float16)Wh[g * 64 + k];
  }
  for (int e = lane; e < 1024; e += 32) { hf[e] = 0.f; h16[e] = (_Float16)0.f; }
  __syncthreads();
  for (int t = 0; t < 128; ++t) {
    for (int e = lane; e < 1024; e += 32) {
      int rr = e >> 6, c = e & 63;
      x16[(c >> 5) * 512 + a_swz(rr, c & 31)] =
          (_Float16)hin[(((size_t)(rb + rr)) * 128 + t) * 64 + c];
    }
    __syncthreads();
    int n = lane & 15, half = lane >> 4;
    for (int g = 0; g < 12; ++g) {
      v8f ax = {}, ah = {};
#pragma unroll
      for (int kt = 0; kt < 2; ++kt) {
        v16h a1 = *(const v16h*)(x16 + kt * 512 + lane * 16);
        v16h a2 = *(const v16h*)(h16 + kt * 512 + lane * 16);
        v16h b1 = *(const v16h*)(Bx_sw + (g * 2 + kt) * 512 + lane * 16);
        v16h b2 = *(const v16h*)(Bh_sw + (g * 2 + kt) * 512 + lane * 16);
        ax = wmma_f16(a1, b1, ax);
        ah = wmma_f16(a2, b2, ah);
      }
#pragma unroll
      for (int r = 0; r < 8; ++r) {
        int mm = half * 8 + r;
        gx_s[mm * 192 + g * 16 + n] = ax[r];
        gh_s[mm * 192 + g * 16 + n] = ah[r];
      }
    }
    __syncthreads();
    for (int e = lane; e < 1024; e += 32) {
      int rr = e >> 6, c = e & 63;
      float xr_ = gx_s[rr * 192 + c] + bx[c];
      float xz_ = gx_s[rr * 192 + 64 + c] + bx[64 + c];
      float xn_ = gx_s[rr * 192 + 128 + c] + bx[128 + c];
      float hr_ = gh_s[rr * 192 + c] + bh[c];
      float hz_ = gh_s[rr * 192 + 64 + c] + bh[64 + c];
      float hn_ = gh_s[rr * 192 + 128 + c] + bh[128 + c];
      float rg = 1.f / (1.f + expf(-(xr_ + hr_)));
      float zg = 1.f / (1.f + expf(-(xz_ + hz_)));
      float ng = tanhf(xn_ + rg * hn_);
      float hnew = (1.f - zg) * ng + zg * hf[e];
      hf[e] = hnew;
      h16[(c >> 5) * 512 + a_swz(rr, c & 31)] = (_Float16)hnew;
      hs[(((size_t)(rb + rr)) * 128 + t) * 64 + c] = hnew;
    }
    __syncthreads();
  }
}

// ---- K9: y = hs @ out_W^T + out_b ----------------------------------------
__global__ void k_out(const float* __restrict__ hsin, const float* __restrict__ outW,
                      const float* __restrict__ outB, float* __restrict__ y) {
  __shared__ __align__(32) _Float16 Wsw[4 * 2 * 512];  // swizzled out_W^T
  __shared__ __align__(32) _Float16 As2[2 * 512];
  int mt = blockIdx.x;
  for (int e = threadIdx.x; e < 4096; e += blockDim.x) {
    int nt = e >> 10, rem = e & 1023, kt = rem >> 9, le = rem & 511;
    int l2 = le >> 4, j = le & 15;
    int n = l2 & 15, h2 = l2 >> 4;
    int k = kt * 32 + h2 * 16 + j;
    Wsw[e] = (_Float16)outW[(size_t)(nt * 16 + n) * 64 + k];
  }
  const float* Ar = hsin + (size_t)mt * 16 * 64;
  for (int e = threadIdx.x; e < 1024; e += blockDim.x) {
    int row = e >> 6, k = e & 63;
    As2[(k >> 5) * 512 + a_swz(row, k & 31)] = (_Float16)Ar[row * 64 + k];
  }
  __syncthreads();
  int ntile = threadIdx.x >> 5, lane = threadIdx.x & 31;
  v8f acc = {};
#pragma unroll
  for (int kt = 0; kt < 2; ++kt) {
    v16h a = *(const v16h*)(As2 + kt * 512 + lane * 16);
    v16h b = *(const v16h*)(Wsw + (ntile * 2 + kt) * 512 + lane * 16);
    acc = wmma_f16(a, b, acc);
  }
  int n = lane & 15, half = lane >> 4;
  float bv = outB[ntile * 16 + n];
  float* drow = y + (size_t)mt * 16 * 64 + ntile * 16;
#pragma unroll
  for (int r = 0; r < 8; ++r) drow[(size_t)(half * 8 + r) * 64 + n] = acc[r] + bv;
}

extern "C" void kernel_launch(void* const* d_in, const int* in_sizes, int n_in,
                              void* d_out, int out_size, void* d_ws, size_t ws_size,
                              hipStream_t stream) {
  (void)in_sizes; (void)n_in; (void)out_size; (void)ws_size;
  const float* x       = (const float*)d_in[0];
  const float* gridv   = (const float*)d_in[1];
  const float* pole_re = (const float*)d_in[2];
  const float* pole_im = (const float*)d_in[3];
  const float* res_re  = (const float*)d_in[4];
  const float* res_im  = (const float*)d_in[5];
  const float* conv_w  = (const float*)d_in[6];
  const float* conv_b  = (const float*)d_in[7];
  const float* gru_Wx  = (const float*)d_in[8];
  const float* gru_Wh  = (const float*)d_in[9];
  const float* gru_bx  = (const float*)d_in[10];
  const float* gru_bh  = (const float*)d_in[11];
  const float* out_W   = (const float*)d_in[12];
  const float* out_b   = (const float*)d_in[13];

  const size_t NBIG = (size_t)ROWS * NXD;              // 16,777,216 elements
  float* xs    = (float*)d_ws;
  float* xsn   = xs + NBIG;
  _Float16* Csw = (_Float16*)(xsn + NBIG);
  _Float16* Ssw = Csw + (size_t)NXD * NXD;
  _Float16* afr = Ssw + (size_t)NXD * NXD;             // alpha re, swizzled f16
  _Float16* afi = afr + NBIG;                          // alpha im, swizzled f16
  float* o1re  = (float*)(afi + NBIG);
  float* o1im  = o1re + NBIG;
  float* o2re  = o1im + NBIG;
  float* o2im  = o2re + (size_t)BT * 1024;
  _Float16* o2swr = (_Float16*)(o2im + (size_t)BT * 1024);
  _Float16* o2swi = o2swr + (size_t)BT * 1024;
  float* x12   = (float*)(o2swi + (size_t)BT * 1024);
  float* hgru  = x12 + NBIG;
  float* hs    = hgru + NBIG;
  float* y     = (float*)d_out;

  k_pack_norm<<<ROWS, 256, 0, stream>>>(x, xs, xsn);
  k_gen_dft<<<(NXD * NXD + 255) / 256, 256, 0, stream>>>(Csw, Ssw);
  // FFT: alpha_re = xsn @ C ; alpha_im = -(xsn @ S), emitted as swizzled f16
  k_gemm<1><<<dim3(ROWS / 16, 4), 256, 0, stream>>>(xsn, Csw, nullptr, afr, 1.f, 0.f);
  k_gemm<1><<<dim3(ROWS / 16, 4), 256, 0, stream>>>(xsn, Ssw, nullptr, afi, -1.f, 0.f);
  k_or1<<<512, 256, 0, stream>>>(afr, afi, pole_re, pole_im, res_re, res_im, gridv, o1re, o1im);
  k_or2<<<dim3(32, 8), 256, 0, stream>>>(afr, afi, pole_re, pole_im, res_re, res_im, gridv, o2re, o2im);
  k_prep_or2<<<(BT * 1024) / 256, 256, 0, stream>>>(o2re, o2im, o2swr, o2swi);
  // x1 = Re(ifft(or1)) = (or1_re@C - or1_im@S)/N  -> x12 (f32)
  k_gemm<0><<<dim3(ROWS / 16, 4), 256, 0, stream>>>(o1re, Csw, x12, nullptr, 1.f / 512.f, 0.f);
  k_gemm<0><<<dim3(ROWS / 16, 4), 256, 0, stream>>>(o1im, Ssw, x12, nullptr, -1.f / 512.f, 1.f);
  k_x2<<<dim3(64, 32), 256, 0, stream>>>(o2swr, o2swi, pole_re, pole_im, gridv, x12);
  k_ncs<<<ROWS, 256, 0, stream>>>(x12, xs, conv_w, conv_b, hgru);
  k_gru<<<128, 32, 0, stream>>>(hgru, gru_Wx, gru_Wh, gru_bx, gru_bh, hs);
  k_out<<<(BDIM * NXD * NTD) / 16, 128, 0, stream>>>(hs, out_W, out_b, y);
}
